// PMA_49263274885652
// MI455X (gfx1250) — matrix operand order repaired and checked by reading, not compiled, not run on hardware
//
#include <hip/hip_runtime.h>
#include <hip/hip_bf16.h>

// ---------------------------------------------------------------------------
// Pooled multi-head attention, algebraically reduced:
//   scores[m,n] = (A'[m,:] . xs[n,:]) * rscale[n] + sbias[m]     (bf16 WMMA)
//   p'[m,n]    = exp(score - max_n) * rscale[n]
//   B[m,c]     = sum_n p'[m,n] * xs[n,c]                          (bf16 WMMA)
//   out        = sanitize(mean_s (Wout (Wv (w_nkv*B/denom) + bv)) + bout)
// m = s*4 + h  (72 real rows, padded to 96 = 6 WMMA M-tiles)
// ---------------------------------------------------------------------------

typedef __attribute__((ext_vector_type(16))) __bf16 bf16x16;
typedef __attribute__((ext_vector_type(8)))  __bf16 bf16x8;
typedef __attribute__((ext_vector_type(8)))  float  f32x8;

#define CDIM 1024
#define SQ   18
#define MPAD 96          // 72 real (s,h) rows padded to 6 tiles of 16
#define ROWS 16          // x rows per chunk in pass1/pass2
#define LDSW 1032        // padded LDS row stride (bf16 elems)
#define P3_ROWS 4000     // rows per pass-3 K-strip
#define P3_KSTEPS (P3_ROWS/32)

__device__ __forceinline__ float san(float v) {
    if (__builtin_isnan(v)) return 0.f;
    return fminf(fmaxf(v, -10000.f), 10000.f);
}
__device__ __forceinline__ unsigned enc_f(float x) {
    unsigned u = __float_as_uint(x);
    return (u & 0x80000000u) ? ~u : (u | 0x80000000u);
}
__device__ __forceinline__ float dec_f(unsigned e) {
    unsigned u = (e & 0x80000000u) ? (e & 0x7FFFFFFFu) : ~e;
    return __uint_as_float(u);
}
__device__ __forceinline__ unsigned short f2bf(float f) {
    return __builtin_bit_cast(unsigned short, (__bf16)f);
}
__device__ __forceinline__ __bf16 u2bf(unsigned short u) {
    return __builtin_bit_cast(__bf16, u);
}
__device__ __forceinline__ bf16x8 ld8(const unsigned short* p) {
    return *reinterpret_cast<const bf16x8*>(p);
}
__device__ __forceinline__ bf16x16 frag_cat(bf16x8 lo, bf16x8 hi) {
    bf16x16 r;
#pragma unroll
    for (int i = 0; i < 8; ++i) { r[i] = lo[i]; r[i + 8] = hi[i]; }
    return r;
}
__device__ __forceinline__ f32x8 wmma_bf16(bf16x16 a, bf16x16 b, f32x8 c) {
    return __builtin_amdgcn_wmma_f32_16x16x32_bf16(false, a, false, b,
                                                   (short)0, c, false, false);
}

// ---------------- init: zero accumulators + A' pad rows --------------------
__global__ __launch_bounds__(256) void k_init(float* __restrict__ Bacc,
                                              float* __restrict__ denom,
                                              unsigned* __restrict__ gmax,
                                              float* __restrict__ sbias,
                                              unsigned short* __restrict__ Ab) {
    int i = blockIdx.x * 256 + threadIdx.x;
    if (i < MPAD * CDIM) Bacc[i] = 0.f;
    if (i < MPAD) { denom[i] = 0.f; gmax[i] = 0u; sbias[i] = 0.f; }
    if (i < 24 * CDIM) Ab[72 * CDIM + i] = 0;   // zero padded A' rows
}

// ---------------- q row scales (1 block) -----------------------------------
__global__ __launch_bounds__(256) void k_qscale(const float* __restrict__ seeds,
                                                float* __restrict__ qscale) {
    __shared__ float part[8];
    int t = threadIdx.x;
    for (int s = 0; s < SQ; ++s) {
        float ps = 0.f;
        for (int c = t; c < CDIM; c += 256) {
            float v = san(seeds[s * CDIM + c]);
            ps += v * v;
        }
        for (int m = 16; m >= 1; m >>= 1) ps += __shfl_xor(ps, m);
        if ((t & 31) == 0) part[t >> 5] = ps;
        __syncthreads();
        if (t == 0) {
            float tot = 0.f;
            for (int w = 0; w < 8; ++w) tot += part[w];
            qscale[s] = 1.f / (sqrtf(tot) * (1.f / 32.f) + 1e-8f);
        }
        __syncthreads();
    }
}

// ---------------- qp = (rmsnorm(seeds) @ Wq^T + bq) / 16 -------------------
__global__ __launch_bounds__(256) void k_qp(const float* __restrict__ seeds,
                                            const float* __restrict__ w_nq,
                                            const float* __restrict__ qscale,
                                            const float* __restrict__ Wfull,
                                            const float* __restrict__ bfull,
                                            float* __restrict__ qp) {
    __shared__ float qn[CDIM];
    int s = blockIdx.x, jb = blockIdx.y, t = threadIdx.x;
    float qs = qscale[s];
    for (int c = t; c < CDIM; c += 256)
        qn[c] = w_nq[c] * san(seeds[s * CDIM + c]) * qs;
    __syncthreads();
    int j = jb * 256 + t;
    const float* Wr = Wfull + (long)j * CDIM;
    float acc = 0.f;
#pragma unroll 4
    for (int c = 0; c < CDIM; ++c) acc += Wr[c] * qn[c];
    qp[s * CDIM + j] = (acc + bfull[j]) * 0.0625f;
}

// ---------------- A'[m,c] = w_nkv[c] * sum_d qp[s,h*256+d]*Wk[h*256+d,c] ---
__global__ __launch_bounds__(256) void k_A(const float* __restrict__ qp,
                                           const float* __restrict__ Wfull,
                                           const float* __restrict__ bfull,
                                           const float* __restrict__ w_nkv,
                                           unsigned short* __restrict__ Ab,
                                           float* __restrict__ sbias) {
    __shared__ float qh[256];
    int m = blockIdx.x;           // 0..71
    int s = m >> 2, h = m & 3, jb = blockIdx.y, t = threadIdx.x;
    qh[t] = qp[s * CDIM + h * 256 + t];
    __syncthreads();
    int c = jb * 256 + t;
    float acc = 0.f;
#pragma unroll 4
    for (int d = 0; d < 256; ++d)
        acc += qh[d] * Wfull[(long)(CDIM + h * 256 + d) * CDIM + c];
    Ab[m * CDIM + c] = f2bf(w_nkv[c] * acc);
    if (jb == 0 && t == 0) {
        float sb = 0.f;
        for (int d = 0; d < 256; ++d) sb += qh[d] * bfull[CDIM + h * 256 + d];
        sbias[m] = sb;
    }
}

// ---------------- pass1: sanitize x -> xs(bf16), rscale; scores -> gmax ----
__global__ __launch_bounds__(256) void k_pass1(const float* __restrict__ x,
                                               const unsigned short* __restrict__ Ab,
                                               const float* __restrict__ sbias,
                                               unsigned short* __restrict__ xsw,
                                               float* __restrict__ rscale,
                                               unsigned* __restrict__ gmax) {
    __shared__ __align__(16) unsigned short xls[ROWS * LDSW];
    __shared__ float ssq[ROWS];
    int tid = threadIdx.x;
    long r0 = (long)blockIdx.x * ROWS;
    if (tid < ROWS) ssq[tid] = 0.f;
    __syncthreads();
    for (int k = 0; k < ROWS; ++k) {
        float4 v4 = ((const float4*)(x + (r0 + k) * CDIM))[tid];
        float a = san(v4.x), b = san(v4.y), c = san(v4.z), d = san(v4.w);
        float ps = a * a + b * b + c * c + d * d;
        for (int m = 16; m >= 1; m >>= 1) ps += __shfl_xor(ps, m);
        if ((tid & 31) == 0) atomicAdd(&ssq[k], ps);
        unsigned lo = (unsigned)f2bf(a) | ((unsigned)f2bf(b) << 16);
        unsigned hi = (unsigned)f2bf(c) | ((unsigned)f2bf(d) << 16);
        ((uint2*)(xsw + (r0 + k) * CDIM))[tid] = make_uint2(lo, hi);
        *reinterpret_cast<uint2*>(&xls[k * LDSW + 4 * tid]) = make_uint2(lo, hi);
    }
    __syncthreads();
    if (tid < ROWS) {
        float rs = 1.f / (sqrtf(ssq[tid]) * (1.f / 32.f) + 1e-8f);
        ssq[tid] = rs;
        rscale[r0 + tid] = rs;
    }
    __syncthreads();

    int wid = tid >> 5;
    if (wid < 6) {                         // 6 M-tiles of 16 (96 rows)
        int lane = tid & 31, half = lane >> 4, l = lane & 15;
        f32x8 acc = {0.f, 0.f, 0.f, 0.f, 0.f, 0.f, 0.f, 0.f};
        const unsigned short* Arow = Ab + (wid * 16 + l) * CDIM;
        const unsigned short* Brow = &xls[l * LDSW];
        for (int kb = 0; kb < CDIM; kb += 32) {
            bf16x16 af = frag_cat(ld8(Arow + kb + half * 8),
                                  ld8(Arow + kb + 16 + half * 8));
            bf16x16 bf = frag_cat(ld8(Brow + kb + half * 8),
                                  ld8(Brow + kb + 16 + half * 8));
            acc = wmma_bf16(af, bf, acc);
        }
        float rs = ssq[l];
#pragma unroll
        for (int r = 0; r < 8; ++r) {
            int m = wid * 16 + half * 8 + r;
            float sc = acc[r] * rs + sbias[m];
            for (int msk = 8; msk >= 1; msk >>= 1)
                sc = fmaxf(sc, __shfl_xor(sc, msk));
            if (l == 0) atomicMax(&gmax[m], enc_f(sc));
        }
    }
}

// ---------------- pass2: scores again -> p' (bf16), denom ------------------
__global__ __launch_bounds__(192) void k_pass2(const unsigned short* __restrict__ xsw,
                                               const unsigned short* __restrict__ Ab,
                                               const float* __restrict__ sbias,
                                               const float* __restrict__ rscale,
                                               const unsigned* __restrict__ gmax,
                                               unsigned short* __restrict__ Pw,
                                               float* __restrict__ denom, long N) {
    int tid = threadIdx.x;
    int wid = tid >> 5, lane = tid & 31, half = lane >> 4, l = lane & 15;
    long r0 = (long)blockIdx.x * ROWS;
    f32x8 acc = {0.f, 0.f, 0.f, 0.f, 0.f, 0.f, 0.f, 0.f};
    const unsigned short* Arow = Ab + (wid * 16 + l) * CDIM;
    const unsigned short* Brow = xsw + (r0 + l) * CDIM;
    for (int kb = 0; kb < CDIM; kb += 32) {
        bf16x16 af = frag_cat(ld8(Arow + kb + half * 8),
                              ld8(Arow + kb + 16 + half * 8));
        bf16x16 bf = frag_cat(ld8(Brow + kb + half * 8),
                              ld8(Brow + kb + 16 + half * 8));
        acc = wmma_bf16(af, bf, acc);
    }
    float rs = rscale[r0 + l];
#pragma unroll
    for (int r = 0; r < 8; ++r) {
        int m = wid * 16 + half * 8 + r;
        float sc = acc[r] * rs + sbias[m];
        float p = __expf(sc - dec_f(gmax[m]));
        Pw[(long)m * N + r0 + l] = f2bf(p * rs);
        float s = p;
        for (int msk = 8; msk >= 1; msk >>= 1) s += __shfl_xor(s, msk);
        if (l == 0) atomicAdd(&denom[m], s);
    }
}

// ---------------- pass3: Bacc[m,c] += sum_n p'[m,n]*xs[n,c] ----------------
__global__ __launch_bounds__(256) void k_pass3(const unsigned short* __restrict__ xsw,
                                               const unsigned short* __restrict__ Pw,
                                               float* __restrict__ Bacc, long N) {
    int tid = threadIdx.x;
    int wid = tid >> 5, lane = tid & 31, half = lane >> 4, l = lane & 15;
    int ctile = blockIdx.x * 8 + wid;     // 64 c-tiles, 8 per block
    int c = ctile * 16 + l;
    long nbase = (long)blockIdx.y * P3_ROWS;
    f32x8 acc[6];
#pragma unroll
    for (int mt = 0; mt < 6; ++mt)
#pragma unroll
        for (int r = 0; r < 8; ++r) acc[mt][r] = 0.f;

    for (int t = 0; t < P3_KSTEPS; ++t) {
        long n0 = nbase + 32 * t;
        bf16x16 bf;
#pragma unroll
        for (int e = 0; e < 16; ++e)
            bf[e] = u2bf(xsw[(n0 + half * 16 + e) * CDIM + c]);
#pragma unroll
        for (int mt = 0; mt < 6; ++mt) {
            const unsigned short* Pr = Pw + (long)(mt * 16 + l) * N + n0 + half * 16;
            bf16x16 af = frag_cat(ld8(Pr), ld8(Pr + 8));
            acc[mt] = wmma_bf16(af, bf, acc[mt]);
        }
    }
#pragma unroll
    for (int mt = 0; mt < 6; ++mt)
#pragma unroll
        for (int r = 0; r < 8; ++r)
            atomicAdd(&Bacc[(mt * 16 + half * 8 + r) * CDIM + c], acc[mt][r]);
}

// ---------------- o[s,j] = Wv[j,:].(w_nkv*Bacc[m]/denom) + bv[j] -----------
__global__ __launch_bounds__(256) void k_o(const float* __restrict__ Bacc,
                                           const float* __restrict__ denom,
                                           const float* __restrict__ w_nkv,
                                           const float* __restrict__ Wfull,
                                           const float* __restrict__ bfull,
                                           float* __restrict__ o) {
    __shared__ float bn[CDIM];
    int s = blockIdx.x, h = blockIdx.y, t = threadIdx.x;
    int m = s * 4 + h;
    float inv = 1.f / denom[m];
    for (int c = t; c < CDIM; c += 256)
        bn[c] = w_nkv[c] * Bacc[m * CDIM + c] * inv;
    __syncthreads();
    int j = h * 256 + t;
    const float* Wr = Wfull + (long)(2 * CDIM + j) * CDIM;
    float acc = 0.f;
#pragma unroll 4
    for (int c = 0; c < CDIM; ++c) acc += Wr[c] * bn[c];
    o[s * CDIM + j] = acc + bfull[2 * CDIM + j];
}

__global__ __launch_bounds__(256) void k_om(const float* __restrict__ o,
                                            float* __restrict__ om) {
    int j = blockIdx.x * 256 + threadIdx.x;
    float a = 0.f;
    for (int s = 0; s < SQ; ++s) a += o[s * CDIM + j];
    om[j] = a * (1.f / (float)SQ);
}

__global__ __launch_bounds__(256) void k_y(const float* __restrict__ om,
                                           const float* __restrict__ Wout,
                                           const float* __restrict__ bout,
                                           float* __restrict__ out) {
    __shared__ float oml[CDIM];
    int t = threadIdx.x;
    for (int c = t; c < CDIM; c += 256) oml[c] = om[c];
    __syncthreads();
    int i = blockIdx.x * 256 + t;
    const float* Wr = Wout + (long)i * CDIM;
    float acc = 0.f;
#pragma unroll 4
    for (int j = 0; j < CDIM; ++j) acc += Wr[j] * oml[j];
    out[i] = san(acc + bout[i]);
}

// ---------------------------------------------------------------------------
extern "C" void kernel_launch(void* const* d_in, const int* in_sizes, int n_in,
                              void* d_out, int out_size, void* d_ws, size_t ws_size,
                              hipStream_t stream) {
    const float* x     = (const float*)d_in[0];
    const float* seeds = (const float*)d_in[1];
    const float* w_nq  = (const float*)d_in[2];
    const float* w_nkv = (const float*)d_in[3];
    const float* Wfull = (const float*)d_in[4];   // (3C, C)
    const float* bfull = (const float*)d_in[5];   // (3C)
    const float* Wout  = (const float*)d_in[6];   // (C, C)
    const float* bout  = (const float*)d_in[7];   // (C)
    float* out = (float*)d_out;

    const long N = (long)in_sizes[0] / CDIM;      // 100000

    char* w = (char*)d_ws;
    size_t off = 0;
    auto take = [&](size_t bytes) {
        size_t o = off;
        off = (off + bytes + 255) & ~(size_t)255;
        return o;
    };
    unsigned short* xs   = (unsigned short*)(w + take((size_t)N * CDIM * 2));
    float*          rsc  = (float*)(w + take((size_t)N * 4));
    float*          qsc  = (float*)(w + take(SQ * 4));
    float*          qp   = (float*)(w + take(SQ * CDIM * 4));
    unsigned short* Ab   = (unsigned short*)(w + take(MPAD * CDIM * 2));
    float*          sbias= (float*)(w + take(MPAD * 4));
    unsigned*       gmax = (unsigned*)(w + take(MPAD * 4));
    float*          den  = (float*)(w + take(MPAD * 4));
    unsigned short* Pw   = (unsigned short*)(w + take((size_t)MPAD * N * 2));
    float*          Bacc = (float*)(w + take(MPAD * CDIM * 4));
    float*          obuf = (float*)(w + take(SQ * CDIM * 4));
    float*          om   = (float*)(w + take(CDIM * 4));
    (void)ws_size; (void)n_in; (void)out_size;

    k_init  <<<(MPAD * CDIM + 255) / 256, 256, 0, stream>>>(Bacc, den, gmax, sbias, Ab);
    k_qscale<<<1, 256, 0, stream>>>(seeds, qsc);
    k_qp    <<<dim3(SQ, 4), 256, 0, stream>>>(seeds, w_nq, qsc, Wfull, bfull, qp);
    k_A     <<<dim3(72, 4), 256, 0, stream>>>(qp, Wfull, bfull, w_nkv, Ab, sbias);
    k_pass1 <<<(int)(N / ROWS), 256, 0, stream>>>(x, Ab, sbias, xs, rsc, gmax);
    k_pass2 <<<(int)(N / ROWS), 192, 0, stream>>>(xs, Ab, sbias, rsc, gmax, Pw, den, N);
    k_pass3 <<<dim3(8, (int)(N / P3_ROWS)), 256, 0, stream>>>(xs, Pw, Bacc, N);
    k_o     <<<dim3(SQ, 4), 256, 0, stream>>>(Bacc, den, w_nkv, Wfull, bfull, obuf);
    k_om    <<<4, 256, 0, stream>>>(obuf, om);
    k_y     <<<4, 256, 0, stream>>>(om, Wout, bout, out);
}
